// SelfAttentionBlock_11132555231928
// MI455X (gfx1250) — matrix operand order repaired and checked by reading, not compiled, not run on hardware
//
#include <hip/hip_runtime.h>

// ---------------------------------------------------------------------------
// SelfAttentionBlock for MI455X (gfx1250, wave32, WMMA f16->f32)
//   B=8, C=256, H=W=64 (N=4096), GROUPS=32
// Pipeline:
//   1) gn_kernel   : GroupNorm, f32 [B,C,N] -> f16 hT [B,N,C]
//   2) cvt_kernel  : f32 weights -> f16 (x4)
//   3) qkv_kernel  : Q[B,N,C], K[B,N,C], Vt[B,C,N]  (WMMA)
//   4) attn_kernel : flash attention with double-buffered async K/V LDS
//                    staging (ASYNCcnt) + WMMA + LDS P-transpose
//   5) out_kernel  : Wo proj + bias + residual -> f32 out [B,C,H,W] (WMMA)
// ---------------------------------------------------------------------------

constexpr int kB      = 8;
constexpr int kC      = 256;
constexpr int kN      = 4096;          // 64*64 tokens
constexpr int kGroups = 32;
constexpr int kCpg    = kC / kGroups;  // 8
constexpr float kEps   = 1e-5f;
constexpr float kScale = 0.0625f;      // 1/sqrt(256)

// LDS staging geometry for attention (padded rows -> conflict-free ds_load_b128)
constexpr int kKRow = 264;             // 256 + 8 halves pad
constexpr int kVRow = 40;              // 32 + 8 halves pad
constexpr int kPRow = 40;              // 32 + 8 halves pad
constexpr int kKBuf = 32 * kKRow;      // halves per K buffer
constexpr int kVBuf = kC * kVRow;      // halves per V buffer
constexpr int kPBuf = 16 * kPRow;      // halves per wave P buffer
constexpr size_t kAttnLds =
    (size_t)(2 * kKBuf + 2 * kVBuf + 8 * kPBuf) * 2;  // bytes

typedef _Float16 half_t;
typedef __attribute__((ext_vector_type(16))) _Float16 v16h;
typedef __attribute__((ext_vector_type(8)))  _Float16 v8h;
typedef __attribute__((ext_vector_type(8)))  float    v8f;
typedef __attribute__((ext_vector_type(4)))  int      v4i;

union HV16 { v16h v; v8h h[2]; };

__device__ __forceinline__ v8f wmma16(v16h a, v16h b, v8f c) {
  return __builtin_amdgcn_wmma_f32_16x16x32_f16(false, a, false, b, (short)0, c,
                                                false, false);
}

__device__ __forceinline__ v8f zero8() {
  v8f z = {0.f, 0.f, 0.f, 0.f, 0.f, 0.f, 0.f, 0.f};
  return z;
}

// A operand: 16(M) x 32(K) tile from row-major f16 [rows x ld], contiguous K.
__device__ __forceinline__ v16h load_a(const half_t* p, int ld, int m0, int k0,
                                       int lane) {
  const half_t* base =
      p + (size_t)(m0 + (lane & 15)) * ld + (k0 + ((lane >> 4) << 4));
  HV16 u;
  u.h[0] = *(const v8h*)(base);
  u.h[1] = *(const v8h*)(base + 8);
  return u.v;
}

// B operand: 32(K) x 16(col) tile where B[k][col] = src[col][k];
// src row-major [cols x ld], contiguous K. Lane = col (mod 16);
// holds 16 contiguous k at k0 + 16*lh.
__device__ __forceinline__ v16h load_b(const half_t* p, int ld, int c0, int k0,
                                       int lane) {
  const half_t* base =
      p + (size_t)(c0 + (lane & 15)) * ld + (k0 + ((lane >> 4) << 4));
  HV16 u;
  u.h[0] = *(const v8h*)(base);
  u.h[1] = *(const v8h*)(base + 8);
  return u.v;
}

// B operand from an LDS row (16 contiguous halves at base).
__device__ __forceinline__ v16h lds_b16(const half_t* base) {
  HV16 u;
  u.h[0] = *(const v8h*)(base);
  u.h[1] = *(const v8h*)(base + 8);
  return u.v;
}

// 16-byte global -> LDS copy (async DMA if the toolchain exposes it).
#if __has_builtin(__builtin_amdgcn_global_load_async_to_lds_b128)
typedef v4i __attribute__((address_space(1)))* gp128_t;  // global v4i*
typedef v4i __attribute__((address_space(3)))* lp128_t;  // LDS v4i*
#endif

__device__ __forceinline__ void async_cp16(const half_t* g, half_t* l) {
#if __has_builtin(__builtin_amdgcn_global_load_async_to_lds_b128)
  __builtin_amdgcn_global_load_async_to_lds_b128((gp128_t)g, (lp128_t)l, 0, 0);
#else
  *(v8h*)l = *(const v8h*)g;  // fallback: global_load + ds_store
#endif
}

__device__ __forceinline__ void wait_async_prev() {
  // <=8 outstanding: drain previous tile's 8 copies, keep current 8 in flight
#if __has_builtin(__builtin_amdgcn_s_wait_asynccnt)
  __builtin_amdgcn_s_wait_asynccnt(8);
#else
  __asm__ __volatile__("s_wait_asynccnt 0x8" ::: "memory");
#endif
}

__device__ __forceinline__ void wait_async_all() {
#if __has_builtin(__builtin_amdgcn_s_wait_asynccnt)
  __builtin_amdgcn_s_wait_asynccnt(0);
#else
  __asm__ __volatile__("s_wait_asynccnt 0x0" ::: "memory");
#endif
}

// ------------------------- GroupNorm ---------------------------------------
__global__ __launch_bounds__(256) void gn_kernel(const float* __restrict__ x,
                                                 const float* __restrict__ gnw,
                                                 const float* __restrict__ gnb,
                                                 half_t* __restrict__ hT) {
  int bg = blockIdx.x;
  int bb = bg / kGroups;
  int g  = bg % kGroups;
  int c0 = g * kCpg;
  const float* xp = x + ((size_t)bb * kC + c0) * kN;
  int tid = threadIdx.x;

  float s = 0.f, ss = 0.f;
  for (int i = tid; i < kCpg * kN; i += 256) {
    float v = xp[i];
    s += v;
    ss += v * v;
  }
#pragma unroll
  for (int d = 16; d > 0; d >>= 1) {
    s += __shfl_xor(s, d, 32);
    ss += __shfl_xor(ss, d, 32);
  }
  __shared__ float red[2][8];
  int wv = tid >> 5;
  if ((tid & 31) == 0) { red[0][wv] = s; red[1][wv] = ss; }
  __syncthreads();
  float ts = 0.f, tss = 0.f;
#pragma unroll
  for (int i = 0; i < 8; ++i) { ts += red[0][i]; tss += red[1][i]; }
  const float invn = 1.f / (float)(kCpg * kN);
  float mean = ts * invn;
  float var  = tss * invn - mean * mean;
  float rstd = rsqrtf(var + kEps);

  float gw[kCpg], gb[kCpg];
#pragma unroll
  for (int cc = 0; cc < kCpg; ++cc) {
    float w = gnw[c0 + cc] * rstd;
    gw[cc] = w;
    gb[cc] = gnb[c0 + cc] - mean * w;
  }
  for (int sp = tid; sp < kN; sp += 256) {
    v8h ov;
#pragma unroll
    for (int cc = 0; cc < kCpg; ++cc) {
      ov[cc] = (half_t)(xp[(size_t)cc * kN + sp] * gw[cc] + gb[cc]);
    }
    *(v8h*)(hT + ((size_t)bb * kN + sp) * kC + c0) = ov;
  }
}

// ------------------------- weight f32 -> f16 -------------------------------
__global__ __launch_bounds__(256) void cvt_kernel(const float* __restrict__ w,
                                                  half_t* __restrict__ o,
                                                  int n) {
  int i = blockIdx.x * 256 + threadIdx.x;
  if (i < n) o[i] = (half_t)w[i];
}

// ------------------------- QKV projections ---------------------------------
__global__ __launch_bounds__(256) void qkv_kernel(
    const half_t* __restrict__ hT, const half_t* __restrict__ wqh,
    const float* __restrict__ bq, const half_t* __restrict__ wkh,
    const float* __restrict__ bk, const half_t* __restrict__ wvh,
    const float* __restrict__ bv, half_t* __restrict__ Q,
    half_t* __restrict__ K, half_t* __restrict__ Vt) {
  int lane = threadIdx.x & 31;
  int wid  = blockIdx.x * 8 + (threadIdx.x >> 5);
  int b  = wid >> 10;
  int r  = wid & 1023;
  int n0 = (r >> 3) * 32;
  int o0 = (r & 7) * 32;
  const half_t* hb = hT + (size_t)b * kN * kC;

  v8f aq[2][2], ak[2][2], av[2][2];
#pragma unroll
  for (int i = 0; i < 2; ++i)
#pragma unroll
    for (int j = 0; j < 2; ++j) {
      aq[i][j] = zero8(); ak[i][j] = zero8(); av[i][j] = zero8();
    }

  for (int k0 = 0; k0 < kC; k0 += 32) {
    v16h a0  = load_a(hb,  kC, n0,      k0, lane);
    v16h a1  = load_a(hb,  kC, n0 + 16, k0, lane);
    v16h wa0 = load_a(wvh, kC, o0,      k0, lane);
    v16h wa1 = load_a(wvh, kC, o0 + 16, k0, lane);
    v16h bq0 = load_b(wqh, kC, o0,      k0, lane);
    v16h bq1 = load_b(wqh, kC, o0 + 16, k0, lane);
    v16h bk0 = load_b(wkh, kC, o0,      k0, lane);
    v16h bk1 = load_b(wkh, kC, o0 + 16, k0, lane);
    v16h hb0 = load_b(hb,  kC, n0,      k0, lane);
    v16h hb1 = load_b(hb,  kC, n0 + 16, k0, lane);

    aq[0][0] = wmma16(a0, bq0, aq[0][0]);
    aq[0][1] = wmma16(a0, bq1, aq[0][1]);
    aq[1][0] = wmma16(a1, bq0, aq[1][0]);
    aq[1][1] = wmma16(a1, bq1, aq[1][1]);

    ak[0][0] = wmma16(a0, bk0, ak[0][0]);
    ak[0][1] = wmma16(a0, bk1, ak[0][1]);
    ak[1][0] = wmma16(a1, bk0, ak[1][0]);
    ak[1][1] = wmma16(a1, bk1, ak[1][1]);

    av[0][0] = wmma16(wa0, hb0, av[0][0]);
    av[0][1] = wmma16(wa0, hb1, av[0][1]);
    av[1][0] = wmma16(wa1, hb0, av[1][0]);
    av[1][1] = wmma16(wa1, hb1, av[1][1]);
  }

  int lm = lane & 15, lh = lane >> 4;
#pragma unroll
  for (int i = 0; i < 2; ++i)
#pragma unroll
    for (int j = 0; j < 2; ++j) {
      int oo = o0 + 16 * j + lm;
      float bqv = bq[oo];
      float bkv = bk[oo];
#pragma unroll
      for (int rr = 0; rr < 8; ++rr) {
        size_t nrow = (size_t)b * kN + n0 + 16 * i + 8 * lh + rr;
        Q[nrow * kC + oo] = (half_t)(aq[i][j][rr] + bqv);
        K[nrow * kC + oo] = (half_t)(ak[i][j][rr] + bkv);
      }
    }
#pragma unroll
  for (int i = 0; i < 2; ++i)
#pragma unroll
    for (int j = 0; j < 2; ++j) {
      int nn = n0 + 16 * j + lm;
#pragma unroll
      for (int rr = 0; rr < 8; ++rr) {
        int oo = o0 + 16 * i + 8 * lh + rr;
        Vt[((size_t)b * kC + oo) * kN + nn] = (half_t)(av[i][j][rr] + bv[oo]);
      }
    }
}

// ------------------------- Flash attention ---------------------------------
// Block = 8 waves, one 16-query tile each, same batch & key sequence.
// K/V tiles (32 keys) staged once per block in LDS via async DMA,
// double-buffered so the copy of tile i+1 overlaps compute of tile i.
__device__ __forceinline__ void stage_tiles(const half_t* __restrict__ Kb,
                                            const half_t* __restrict__ Vb,
                                            int m0, half_t* ks, half_t* vs,
                                            int tid) {
  // K tile: 32 rows x 256 halves -> padded rows of kKRow
#pragma unroll
  for (int i = 0; i < 4; ++i) {
    int j   = tid + 256 * i;
    int row = j >> 5;            // 0..31
    int c8  = (j & 31) << 3;     // 0..248
    async_cp16(Kb + (size_t)(m0 + row) * kC + c8, ks + row * kKRow + c8);
  }
  // V tile: 256 channel-rows x 32 halves -> padded rows of kVRow
#pragma unroll
  for (int i = 0; i < 4; ++i) {
    int j   = tid + 256 * i;
    int c   = j >> 2;            // 0..255
    int off = (j & 3) << 3;      // 0,8,16,24
    async_cp16(Vb + (size_t)c * kN + m0 + off, vs + c * kVRow + off);
  }
}

__global__ __launch_bounds__(256) void attn_kernel(const half_t* __restrict__ Q,
                                                   const half_t* __restrict__ K,
                                                   const half_t* __restrict__ Vt,
                                                   half_t* __restrict__ AO) {
  extern __shared__ char smem[];
  half_t* KsBase = (half_t*)smem;                         // 2 * kKBuf
  half_t* VsBase = KsBase + 2 * kKBuf;                    // 2 * kVBuf
  half_t* PBase  = VsBase + 2 * kVBuf;                    // 8 * kPBuf

  int tid  = threadIdx.x;
  int lane = tid & 31;
  int wv   = tid >> 5;
  int wid  = blockIdx.x * 8 + wv;
  int b  = wid >> 8;
  int q0 = (wid & 255) * 16;
  const half_t* Qb = Q + (size_t)b * kN * kC;
  const half_t* Kb = K + (size_t)b * kN * kC;
  const half_t* Vb = Vt + (size_t)b * kC * kN;
  half_t* P = PBase + wv * kPBuf;
  int lm = lane & 15, lh = lane >> 4;

  // Q tile (16 x 256) as 8 A-operands, kept in VGPRs
  v16h qa[8];
#pragma unroll
  for (int kc = 0; kc < 8; ++kc) qa[kc] = load_a(Qb, kC, q0, kc * 32, lane);

  v8f O[16];
#pragma unroll
  for (int t = 0; t < 16; ++t) O[t] = zero8();
  float mrow[8], lrow[8];
#pragma unroll
  for (int rr = 0; rr < 8; ++rr) { mrow[rr] = -3.0e38f; lrow[rr] = 0.f; }

  constexpr int nIter = kN / 32;  // 128
  // prologue: stage tile 0 into buffer 0
  stage_tiles(Kb, Vb, 0, KsBase, VsBase, tid);

  for (int it = 0; it < nIter; ++it) {
    half_t* ks = KsBase + (it & 1) * kKBuf;
    half_t* vs = VsBase + (it & 1) * kVBuf;
    if (it + 1 < nIter) {
      stage_tiles(Kb, Vb, (it + 1) * 32, KsBase + ((it + 1) & 1) * kKBuf,
                  VsBase + ((it + 1) & 1) * kVBuf, tid);
      wait_async_prev();  // tile `it` landed; tile `it+1` still in flight
    } else {
      wait_async_all();
    }
    __syncthreads();  // whole block sees tile `it`

    // S = Q * K^T for this 16x32 key window (two 16x16 D tiles), K from LDS
    v8f s0 = zero8(), s1 = zero8();
#pragma unroll
    for (int kc = 0; kc < 8; ++kc) {
      v16h kb0 = lds_b16(ks + lm * kKRow + kc * 32 + 16 * lh);
      v16h kb1 = lds_b16(ks + (16 + lm) * kKRow + kc * 32 + 16 * lh);
      s0 = wmma16(qa[kc], kb0, s0);
      s1 = wmma16(qa[kc], kb1, s1);
    }
    // online softmax; lane-half maps to query rows 8*lh+rr (matches C/D layout)
    float scl[8];
#pragma unroll
    for (int rr = 0; rr < 8; ++rr) {
      float v0 = s0[rr] * kScale;
      float v1 = s1[rr] * kScale;
      float mx = fmaxf(v0, v1);
#pragma unroll
      for (int d = 1; d < 16; d <<= 1) mx = fmaxf(mx, __shfl_xor(mx, d, 32));
      float mn = fmaxf(mrow[rr], mx);
      float p0 = __expf(v0 - mn);
      float p1 = __expf(v1 - mn);
      float rs = p0 + p1;
#pragma unroll
      for (int d = 1; d < 16; d <<= 1) rs += __shfl_xor(rs, d, 32);
      float sc = __expf(mrow[rr] - mn);
      lrow[rr] = lrow[rr] * sc + rs;
      mrow[rr] = mn;
      scl[rr]  = sc;
      // D-layout -> per-wave LDS P[16 q][32 keys] (padded rows)
      P[(8 * lh + rr) * kPRow + lm]      = (half_t)p0;
      P[(8 * lh + rr) * kPRow + 16 + lm] = (half_t)p1;
    }
#pragma unroll
    for (int t = 0; t < 16; ++t)
#pragma unroll
      for (int rr = 0; rr < 8; ++rr) O[t][rr] *= scl[rr];

    // wave-internal LDS transpose of P: drain the 2-byte scatter stores
    __asm__ __volatile__("s_wait_dscnt 0x0" ::: "memory");
    HV16 u;
    const half_t* pb = P + lm * kPRow + 16 * lh;
    u.h[0] = *(const v8h*)(pb);
    u.h[1] = *(const v8h*)(pb + 8);
    v16h pa = u.v;

    // O += P * V over 16 c-tiles; V rows read from LDS
#pragma unroll
    for (int ct = 0; ct < 16; ++ct) {
      v16h vb = lds_b16(vs + (ct * 16 + lm) * kVRow + 16 * lh);
      O[ct] = wmma16(pa, vb, O[ct]);
    }
    __syncthreads();  // all waves done reading tile `it` before overwrite
  }

  float inv[8];
#pragma unroll
  for (int rr = 0; rr < 8; ++rr) inv[rr] = 1.f / lrow[rr];
#pragma unroll
  for (int ct = 0; ct < 16; ++ct) {
#pragma unroll
    for (int rr = 0; rr < 8; ++rr) {
      size_t nrow = (size_t)b * kN + q0 + 8 * lh + rr;
      AO[nrow * kC + ct * 16 + lm] = (half_t)(O[ct][rr] * inv[rr]);
    }
  }
}

// ------------------------- Output projection + residual --------------------
__global__ __launch_bounds__(256) void out_kernel(const half_t* __restrict__ AO,
                                                  const half_t* __restrict__ woh,
                                                  const float* __restrict__ bo,
                                                  const float* __restrict__ x,
                                                  float* __restrict__ out) {
  int lane = threadIdx.x & 31;
  int wid  = blockIdx.x * 8 + (threadIdx.x >> 5);
  int b  = wid >> 10;
  int r  = wid & 1023;
  int o0 = (r >> 7) * 32;
  int n0 = (r & 127) * 32;
  const half_t* ab = AO + (size_t)b * kN * kC;

  v8f acc[2][2];
#pragma unroll
  for (int i = 0; i < 2; ++i)
#pragma unroll
    for (int j = 0; j < 2; ++j) acc[i][j] = zero8();

  for (int k0 = 0; k0 < kC; k0 += 32) {
    v16h a0 = load_a(woh, kC, o0,      k0, lane);
    v16h a1 = load_a(woh, kC, o0 + 16, k0, lane);
    v16h b0 = load_b(ab,  kC, n0,      k0, lane);
    v16h b1 = load_b(ab,  kC, n0 + 16, k0, lane);
    acc[0][0] = wmma16(a0, b0, acc[0][0]);
    acc[0][1] = wmma16(a0, b1, acc[0][1]);
    acc[1][0] = wmma16(a1, b0, acc[1][0]);
    acc[1][1] = wmma16(a1, b1, acc[1][1]);
  }
  int lm = lane & 15, lh = lane >> 4;
#pragma unroll
  for (int i = 0; i < 2; ++i)
#pragma unroll
    for (int j = 0; j < 2; ++j) {
      int nn = n0 + 16 * j + lm;
#pragma unroll
      for (int rr = 0; rr < 8; ++rr) {
        int oo = o0 + 16 * i + 8 * lh + rr;
        size_t idx = ((size_t)b * kC + oo) * kN + nn;
        out[idx] = acc[i][j][rr] + bo[oo] + x[idx];
      }
    }
}

// ---------------------------------------------------------------------------
extern "C" void kernel_launch(void* const* d_in, const int* in_sizes, int n_in,
                              void* d_out, int out_size, void* d_ws,
                              size_t ws_size, hipStream_t stream) {
  const float* x   = (const float*)d_in[0];
  const float* gnw = (const float*)d_in[1];
  const float* gnb = (const float*)d_in[2];
  const float* wq  = (const float*)d_in[3];
  const float* bq  = (const float*)d_in[4];
  const float* wk  = (const float*)d_in[5];
  const float* bk  = (const float*)d_in[6];
  const float* wv  = (const float*)d_in[7];
  const float* bv  = (const float*)d_in[8];
  const float* wo  = (const float*)d_in[9];
  const float* bo  = (const float*)d_in[10];
  float* out = (float*)d_out;

  char* ws = (char*)d_ws;
  const size_t eBNC = (size_t)kB * kN * kC;  // 8,388,608 elements
  half_t* hT  = (half_t*)(ws);               // 16 MiB each (f16)
  half_t* Qh  = (half_t*)(ws + 2 * eBNC);
  half_t* Kh  = (half_t*)(ws + 4 * eBNC);
  half_t* Vth = (half_t*)(ws + 6 * eBNC);
  half_t* AOh = (half_t*)(ws + 8 * eBNC);
  half_t* wqh = (half_t*)(ws + 10 * eBNC);
  half_t* wkh = wqh + (size_t)kC * kC;
  half_t* wvh = wkh + (size_t)kC * kC;
  half_t* woh = wvh + (size_t)kC * kC;

  gn_kernel<<<kB * kGroups, 256, 0, stream>>>(x, gnw, gnb, hT);
  cvt_kernel<<<(kC * kC + 255) / 256, 256, 0, stream>>>(wq, wqh, kC * kC);
  cvt_kernel<<<(kC * kC + 255) / 256, 256, 0, stream>>>(wk, wkh, kC * kC);
  cvt_kernel<<<(kC * kC + 255) / 256, 256, 0, stream>>>(wv, wvh, kC * kC);
  cvt_kernel<<<(kC * kC + 255) / 256, 256, 0, stream>>>(wo, woh, kC * kC);
  qkv_kernel<<<1024, 256, 0, stream>>>(hT, wqh, bq, wkh, bk, wvh, bv, Qh, Kh,
                                       Vth);
  attn_kernel<<<256, 256, kAttnLds, stream>>>(Qh, Kh, Vth, AOh);
  out_kernel<<<1024, 256, 0, stream>>>(AOh, woh, bo, x, out);
}
